// ManEmbedder_37306085933536
// MI455X (gfx1250) — compile-verified, matrix operand order checked by reading, not run on hardware
//
#include <hip/hip_runtime.h>
#include <hip/hip_bf16.h>

// ---------------- problem constants (from reference) ----------------
#define H1 128
#define H2 512
#define NG 64
#define KCH 5
#define KTOT (2 * KCH * H1)      // 1280 concatenated K (fwd T0..T4, bwd T0..T4)
#define WLOOP (-1.0f / 3.0f)     // 2/lambda_max - 1, lambda_max = 3
#define WSCALE (2.0f / 3.0f)     // 2/lambda_max

// LDS staging for B: 128 rows x 32 k (64B) + 16B pad -> 80B pitch, double buffered.
#define BROW_PITCH 80
#define BBUF_BYTES (128 * BROW_PITCH)   // 10240
#define LDS_TOTAL  (2 * BBUF_BYTES)     // 20480

typedef __attribute__((ext_vector_type(16))) __bf16 v16bf;
typedef __attribute__((ext_vector_type(8)))  __bf16 v8bf;
typedef __attribute__((ext_vector_type(8)))  float  v8f;
typedef __attribute__((ext_vector_type(4)))  unsigned int v4u;
typedef __attribute__((ext_vector_type(8)))  int v8i;
typedef __attribute__((ext_vector_type(4)))  int v4i;

static __device__ __forceinline__ unsigned short f2bf(float f) {
    unsigned int u = __float_as_uint(f);
    u += 0x7fffu + ((u >> 16) & 1u);   // round to nearest even
    return (unsigned short)(u >> 16);
}
static __device__ __forceinline__ float bf2f(unsigned short h) {
    return __uint_as_float(((unsigned int)h) << 16);
}

// ---------------- graph weight kernels ----------------
__global__ void k_deg(const int* __restrict__ ei, float* __restrict__ deg, int E) {
    int e = blockIdx.x * blockDim.x + threadIdx.x;
    if (e < E) atomicAdd(&deg[ei[e]], 1.0f);
}

__global__ void k_dinv(const float* __restrict__ deg, float* __restrict__ dinv, int N) {
    int i = blockIdx.x * blockDim.x + threadIdx.x;
    if (i < N) {
        float d = deg[i];
        dinv[i] = (d > 0.0f) ? rsqrtf(fmaxf(d, 1e-12f)) : 0.0f;
    }
}

__global__ void k_wedge(const int* __restrict__ ei, const float* __restrict__ dinv,
                        float* __restrict__ wedge, int E) {
    int e = blockIdx.x * blockDim.x + threadIdx.x;
    if (e < E) {
        int r = ei[e], c = ei[E + e];
        wedge[e] = -(dinv[r] * dinv[c]) * WSCALE;
    }
}

// ---------------- layer-1 (scalar features) ----------------
__global__ void k_scat_scalar(const float* __restrict__ t, const int* __restrict__ ei,
                              const float* __restrict__ wedge, float* __restrict__ acc,
                              int E, int rev) {
    int e = blockIdx.x * blockDim.x + threadIdx.x;
    if (e >= E) return;
    int r = ei[e], c = ei[E + e];
    int src = rev ? r : c;
    int dst = rev ? c : r;
    atomicAdd(&acc[dst], wedge[e] * t[src]);
}

__global__ void k_comb_scalar(float* __restrict__ tout, const float* __restrict__ acc,
                              const float* __restrict__ tin, const float* __restrict__ tprev,
                              float alpha, float beta, int N) {
    int i = blockIdx.x * blockDim.x + threadIdx.x;
    if (i < N) tout[i] = alpha * (acc[i] + WLOOP * tin[i]) - beta * tprev[i];
}

// h1 = relu( sum_k Tf_k * W1f[k] + Tb_k * W1b[k] + b1f + b1b ), written bf16 into
// Tcat slot 0 (fwd T0) and slot 5 (bwd T0) — T0 is shared by both directions.
__global__ void k_layer1(const float* __restrict__ x,
                         const float* __restrict__ tf1, const float* __restrict__ tf2,
                         const float* __restrict__ tf3, const float* __restrict__ tf4,
                         const float* __restrict__ tb1, const float* __restrict__ tb2,
                         const float* __restrict__ tb3, const float* __restrict__ tb4,
                         const float* __restrict__ W1f, const float* __restrict__ b1f,
                         const float* __restrict__ W1b, const float* __restrict__ b1b,
                         unsigned short* __restrict__ Tcat, int N) {
    int i = blockIdx.x * blockDim.x + threadIdx.x;
    if (i >= N * H1) return;
    int node = i >> 7, f = i & 127;
    float x0 = x[node];
    float hf = x0        * W1f[f]
             + tf1[node] * W1f[1 * H1 + f]
             + tf2[node] * W1f[2 * H1 + f]
             + tf3[node] * W1f[3 * H1 + f]
             + tf4[node] * W1f[4 * H1 + f];
    float hb = x0        * W1b[f]
             + tb1[node] * W1b[1 * H1 + f]
             + tb2[node] * W1b[2 * H1 + f]
             + tb3[node] * W1b[3 * H1 + f]
             + tb4[node] * W1b[4 * H1 + f];
    float h = fmaxf(hf + hb + b1f[f] + b1b[f], 0.0f);
    unsigned short v = f2bf(h);
    size_t base = (size_t)node * KTOT;
    Tcat[base + f]          = v;   // slot 0: fwd T0
    Tcat[base + 5 * H1 + f] = v;   // slot 5: bwd T0
}

// ---------------- layer-2 (128 features) Chebyshev recursion ----------------
__global__ void k_scat_feat(const unsigned short* __restrict__ Tcat,
                            const int* __restrict__ ei, const float* __restrict__ wedge,
                            float* __restrict__ C, int E, int slot, int rev) {
    int idx = blockIdx.x * blockDim.x + threadIdx.x;
    if (idx >= E * 32) return;
    int e = idx >> 5, q = idx & 31;
    int r = ei[e], c = ei[E + e];
    int src = rev ? r : c;
    int dst = rev ? c : r;
    float w = wedge[e];
    const unsigned short* sp = Tcat + (size_t)src * KTOT + slot * H1 + q * 4;
    uint2 v = *reinterpret_cast<const uint2*>(sp);
    float* dp = C + (size_t)dst * H1 + q * 4;
    atomicAdd(dp + 0, w * bf2f((unsigned short)(v.x & 0xffffu)));
    atomicAdd(dp + 1, w * bf2f((unsigned short)(v.x >> 16)));
    atomicAdd(dp + 2, w * bf2f((unsigned short)(v.y & 0xffffu)));
    atomicAdd(dp + 3, w * bf2f((unsigned short)(v.y >> 16)));
}

__global__ void k_comb_feat(unsigned short* __restrict__ Tcat, const float* __restrict__ C,
                            int N, int slot_out, int s1, int s2, float alpha, float beta) {
    int i = blockIdx.x * blockDim.x + threadIdx.x;
    if (i >= N * H1) return;
    int node = i >> 7, f = i & 127;
    size_t base = (size_t)node * KTOT;
    float t1 = bf2f(Tcat[base + s1 * H1 + f]);
    float t2 = bf2f(Tcat[base + s2 * H1 + f]);
    float o = alpha * (C[i] + WLOOP * t1) - beta * t2;
    Tcat[base + slot_out * H1 + f] = f2bf(o);
}

// Wt[n][kk] = bf16 of concatenated weights, K-major per output column (B^T layout).
__global__ void k_wprep(const float* __restrict__ W2f, const float* __restrict__ W2b,
                        unsigned short* __restrict__ Wt) {
    int i = blockIdx.x * blockDim.x + threadIdx.x;
    if (i >= H2 * KTOT) return;
    int n = i / KTOT, kk = i % KTOT;
    int dir = (kk >= KCH * H1) ? 1 : 0;
    int kr = kk - dir * KCH * H1;
    int k = kr >> 7, r = kr & 127;
    const float* W = dir ? W2b : W2f;
    Wt[(size_t)n * KTOT + kk] = f2bf(W[((size_t)k * H1 + r) * H2 + n]);
}

__global__ void k_counts(const int* __restrict__ batch, float* __restrict__ cnt, int N) {
    int i = blockIdx.x * blockDim.x + threadIdx.x;
    if (i < N) atomicAdd(&cnt[batch[i]], 1.0f);
}

// ---------------- TDM: stage B tile (128 rows x 32 k bf16) into LDS ----------------
// D# per CDNA5 ISA: group0 {count=1, lds_addr, global_addr,
// type=2}; group1 {data_size=2B, pad 16dw->+4dw, tensor dims/strides, tile 32x128}.
// Toolchain uses the 6-arg builtin: (g0, g1, g2, g3, g4, cpol).
static __device__ __forceinline__ void tdm_load_b(const unsigned short* gptr,
                                                  unsigned int lds_off) {
    unsigned long long ga = (unsigned long long)(uintptr_t)gptr;
    v4u g0;
    g0.x = 1u;                                           // count=1, user mode
    g0.y = lds_off;                                      // lds_addr (bytes)
    g0.z = (unsigned int)(ga & 0xffffffffu);             // global_addr[31:0] -> bits[95:64]
    g0.w = (unsigned int)((ga >> 32) & 0x1ffffffu)       // global_addr[56:32]
         | (2u << 30);                                   // type=2 (image)
    v8i g1;
    g1[0] = (1 << 16)      // data_size = 1 -> 2 bytes
          | (1 << 20)      // pad_enable
          | (3 << 22)      // pad_interval = 3 -> every 16 DWORDs (64B row)
          | (3 << 25);     // pad_amount   = 3 -> 4 DWORDs (16B)
    g1[1] = (int)((KTOT & 0xffff) << 16);                // tensor_dim0 = 1280 (lo16)
    g1[2] = (int)0xffff0000u;                            // tdim0 hi=0, tensor_dim1 lo16
    g1[3] = (int)((32u << 16) | 0x7fffu);                // tensor_dim1 hi, tile_dim0=32
    g1[4] = 128;                                         // tile_dim1 = 128, tile_dim2 = 0
    g1[5] = KTOT;                                        // tensor_dim0_stride[31:0] = 1280
    g1[6] = 0;                                           // stride hi, tensor_dim1_stride lo
    g1[7] = 0;
    v4i z4 = {0, 0, 0, 0};
    v8i z8 = {0, 0, 0, 0, 0, 0, 0, 0};
    __builtin_amdgcn_tensor_load_to_lds(g0, g1, z4, z4, z8, 0);
}

// ---------------- fused GEMM (M=100k, K=1280, N=512) + bias + relu + pool ----------------
// 8 waves/block: 4 M-waves x 2 N-waves; wave computes 16x64 via 4 wmma accumulators.
// B tiles are TDM-staged into double-buffered LDS; A streamed from global.
__global__ __launch_bounds__(256) void k_gemm(const unsigned short* __restrict__ Tcat,
                                              const unsigned short* __restrict__ Wt,
                                              const float* __restrict__ b2f,
                                              const float* __restrict__ b2b,
                                              const int* __restrict__ batch,
                                              float* __restrict__ pool, int nN) {
    extern __shared__ char smem[];   // 2 * BBUF_BYTES, dynamic LDS (offset 0)

    const int lane  = threadIdx.x & 31;
    const int wave  = threadIdx.x >> 5;
    const int mw    = wave & 3;
    const int nw    = wave >> 2;
    const int m0    = blockIdx.x * 64 + mw * 16;
    const int nblk  = blockIdx.y * 128;
    const int n0    = nblk + nw * 64;
    const int lrow  = lane & 15;   // M index (A) / N index (B,C)
    const int khalf = lane >> 4;   // which K half-group this lane carries
    const int m     = m0 + lrow;
    // Clamp instead of predicating: invalid rows compute garbage that the
    // epilogue masks out; keeps the hot loop free of exec-mask traffic.
    const int mc    = (m < nN) ? m : (nN - 1);
    const unsigned short* arow = Tcat + (size_t)mc * KTOT;

    v8f acc[4];
    acc[0] = {}; acc[1] = {}; acc[2] = {}; acc[3] = {};

    // Prologue: DMA first B tile, publish to block.
    if (wave == 0) {
        tdm_load_b(Wt + (size_t)nblk * KTOT, 0u);
        __builtin_amdgcn_s_wait_tensorcnt(0);
    }
    __syncthreads();

    int cur = 0;
    for (int k0 = 0; k0 < KTOT; k0 += 32) {
        // Kick DMA for next K-chunk into the other buffer while we compute.
        if (k0 + 32 < KTOT && wave == 0)
            tdm_load_b(Wt + (size_t)nblk * KTOT + (k0 + 32),
                       (unsigned int)((cur ^ 1) * BBUF_BYTES));

        // A fragment: 16x32 bf16, lane lrow = row, two contiguous 16B chunks.
        __builtin_prefetch(arow + k0 + 64, 0, 3);  // global_prefetch next A chunk
        const v8bf alo = *reinterpret_cast<const v8bf*>(arow + k0 + khalf * 8);
        const v8bf ahi = *reinterpret_cast<const v8bf*>(arow + k0 + 16 + khalf * 8);
        v16bf a;
#pragma unroll
        for (int i = 0; i < 8; ++i) { a[i] = alo[i]; a[i + 8] = ahi[i]; }

        // B fragments from LDS (row pitch 80B: TDM pad kills bank conflicts).
        const char* bufc = smem + cur * BBUF_BYTES;
        v16bf b[4];
#pragma unroll
        for (int t = 0; t < 4; ++t) {
            const unsigned short* brow = reinterpret_cast<const unsigned short*>(
                bufc + (nw * 64 + 16 * t + lrow) * BROW_PITCH + khalf * 16);
            const v8bf blo = *reinterpret_cast<const v8bf*>(brow);
            const v8bf bhi = *reinterpret_cast<const v8bf*>(brow + 16);  // +32 bytes
            v16bf bb;
#pragma unroll
            for (int i = 0; i < 8; ++i) { bb[i] = blo[i]; bb[i + 8] = bhi[i]; }
            b[t] = bb;
        }
#pragma unroll
        for (int t = 0; t < 4; ++t)
            acc[t] = __builtin_amdgcn_wmma_f32_16x16x32_bf16(
                false, a, false, b[t], (short)0, acc[t], false, false);

        // Publish next buffer: TDM done (wave 0) then block-wide barrier.
        if (wave == 0) __builtin_amdgcn_s_wait_tensorcnt(0);
        __syncthreads();
        cur ^= 1;
    }

    // Epilogue: bias + relu + mean-pool partial sums (batch is sorted).
    if (m0 >= nN) return;
    const bool uni = (m0 + 15 < nN) && (batch[m0] == batch[m0 + 15]);
#pragma unroll
    for (int t = 0; t < 4; ++t) {
        const int col = n0 + 16 * t + lrow;
        const float bias = b2f[col] + b2b[col];
        if (uni) {
            float s = 0.0f;
#pragma unroll
            for (int r = 0; r < 8; ++r) s += fmaxf(acc[t][r] + bias, 0.0f);
            s += __shfl_xor(s, 16, 32);           // combine the two row-halves
            if (lane < 16) atomicAdd(&pool[batch[m0] * H2 + col], s);
        } else {
#pragma unroll
            for (int r = 0; r < 8; ++r) {
                const int row = m0 + r + 8 * khalf;
                if (row < nN)
                    atomicAdd(&pool[batch[row] * H2 + col], fmaxf(acc[t][r] + bias, 0.0f));
            }
        }
    }
}

__global__ void k_final(const float* __restrict__ pool, const float* __restrict__ cnt,
                        float* __restrict__ out) {
    int i = blockIdx.x * blockDim.x + threadIdx.x;
    if (i >= NG * H2) return;
    int g = i >> 9;
    out[i] = pool[i] / fmaxf(cnt[g], 1.0f);
}

// ---------------- host orchestration ----------------
static inline int cdiv(int a, int b) { return (a + b - 1) / b; }

extern "C" void kernel_launch(void* const* d_in, const int* in_sizes, int n_in,
                              void* d_out, int out_size, void* d_ws, size_t ws_size,
                              hipStream_t stream) {
    (void)n_in; (void)out_size; (void)ws_size;
    const int N = in_sizes[0];          // 100000 nodes
    const int E = in_sizes[1] / 2;      // 400000 edges

    const float* x    = (const float*)d_in[0];
    const int*   ei   = (const int*)d_in[1];
    const int*   bat  = (const int*)d_in[2];
    const float* W1f  = (const float*)d_in[3];
    const float* b1f  = (const float*)d_in[4];
    const float* W1b  = (const float*)d_in[5];
    const float* b1b  = (const float*)d_in[6];
    const float* W2f  = (const float*)d_in[7];
    const float* b2f  = (const float*)d_in[8];
    const float* W2b  = (const float*)d_in[9];
    const float* b2b  = (const float*)d_in[10];
    float* out = (float*)d_out;

    char* p = (char*)d_ws;
    auto alloc = [&](size_t bytes) -> void* {
        void* r = (void*)p;
        p += (bytes + 255) & ~(size_t)255;
        return r;
    };
    float* deg   = (float*)alloc((size_t)N * 4);
    float* dinv  = (float*)alloc((size_t)N * 4);
    float* wedge = (float*)alloc((size_t)E * 4);
    float* tsc[8];
    for (int i = 0; i < 8; ++i) tsc[i] = (float*)alloc((size_t)N * 4);
    float* sacc = (float*)alloc((size_t)N * 4);
    float* C    = (float*)alloc((size_t)N * H1 * 4);
    unsigned short* Tcat = (unsigned short*)alloc((size_t)N * KTOT * 2);
    unsigned short* Wt   = (unsigned short*)alloc((size_t)H2 * KTOT * 2);
    float* pool = (float*)alloc((size_t)NG * H2 * 4);
    float* cnt  = (float*)alloc((size_t)NG * 4);

    const int TB = 256;

    (void)hipMemsetAsync(deg, 0, (size_t)N * 4, stream);
    (void)hipMemsetAsync(pool, 0, (size_t)NG * H2 * 4, stream);
    (void)hipMemsetAsync(cnt, 0, (size_t)NG * 4, stream);

    // Laplacian edge weights (shared by both directions).
    k_deg<<<cdiv(E, TB), TB, 0, stream>>>(ei, deg, E);
    k_dinv<<<cdiv(N, TB), TB, 0, stream>>>(deg, dinv, N);
    k_wedge<<<cdiv(E, TB), TB, 0, stream>>>(ei, dinv, wedge, E);

    // Layer 1: scalar Chebyshev recursion, both directions.
    for (int d = 0; d < 2; ++d) {
        const float* tin = x;
        const float* tprev = x;
        for (int k = 1; k < KCH; ++k) {
            (void)hipMemsetAsync(sacc, 0, (size_t)N * 4, stream);
            k_scat_scalar<<<cdiv(E, TB), TB, 0, stream>>>(tin, ei, wedge, sacc, E, d);
            float alpha = (k == 1) ? 1.0f : 2.0f;
            float beta  = (k == 1) ? 0.0f : 1.0f;
            float* tout = tsc[d * 4 + (k - 1)];
            k_comb_scalar<<<cdiv(N, TB), TB, 0, stream>>>(tout, sacc, tin, tprev, alpha, beta, N);
            tprev = tin;
            tin = tout;
        }
    }
    k_layer1<<<cdiv(N * H1, TB), TB, 0, stream>>>(
        x, tsc[0], tsc[1], tsc[2], tsc[3], tsc[4], tsc[5], tsc[6], tsc[7],
        W1f, b1f, W1b, b1b, Tcat, N);

    // Layer 2: feature Chebyshev recursion into Tcat slots (bf16).
    k_wprep<<<cdiv(H2 * KTOT, TB), TB, 0, stream>>>(W2f, W2b, Wt);
    for (int d = 0; d < 2; ++d) {
        int sb = d * KCH;
        for (int k = 1; k < KCH; ++k) {
            (void)hipMemsetAsync(C, 0, (size_t)N * H1 * 4, stream);
            k_scat_feat<<<cdiv(E * 32, TB), TB, 0, stream>>>(Tcat, ei, wedge, C, E, sb + k - 1, d);
            float alpha = (k == 1) ? 1.0f : 2.0f;
            float beta  = (k == 1) ? 0.0f : 1.0f;
            int s2 = (k == 1) ? sb : (sb + k - 2);
            k_comb_feat<<<cdiv(N * H1, TB), TB, 0, stream>>>(Tcat, C, N, sb + k, sb + k - 1, s2,
                                                             alpha, beta);
        }
    }

    // Pool counts, fused GEMM+bias+relu+pool, final mean.
    k_counts<<<cdiv(N, TB), TB, 0, stream>>>(bat, cnt, N);
    dim3 grid(cdiv(N, 64), H2 / 128);
    k_gemm<<<grid, 256, LDS_TOTAL, stream>>>(Tcat, Wt, b2f, b2b, bat, pool, N);
    k_final<<<cdiv(NG * H2, TB), TB, 0, stream>>>(pool, cnt, out);
}
